// BLAv1_30305289241009
// MI455X (gfx1250) — matrix operand order repaired and checked by reading, not compiled, not run on hardware
//
#include <hip/hip_runtime.h>

// ---------------------------------------------------------------------------
// CDNA5 (gfx1250) BLA pyramid-attention forward.
// All linear layers run through v_wmma_f32_16x16x32_bf16.
// GEMM is register-blocked: one wave computes a 16x64 output strip
// (4 accumulators). A fragment loaded once per K-step; all 4 B fragments are
// preloaded into distinct registers so the 4 WMMAs issue back-to-back under
// partial load waits instead of serializing on a recycled B register set.
// ---------------------------------------------------------------------------

typedef __attribute__((ext_vector_type(16))) __bf16          v16bf;
typedef __attribute__((ext_vector_type(8)))  float           v8f;
typedef __attribute__((ext_vector_type(8)))  unsigned short  us8;
typedef __attribute__((ext_vector_type(16))) unsigned short  us16;

union U16cast { us16 u; v16bf b; };

static __device__ __forceinline__ float b2f(unsigned short h) {
  union { unsigned int u; float f; } c; c.u = ((unsigned int)h) << 16; return c.f;
}
static __device__ __forceinline__ unsigned short f2b(float f) {
  union { float f; unsigned int u; } c; c.f = f;
  unsigned int u = c.u;
  return (unsigned short)((u + 0x7FFFu + ((u >> 16) & 1u)) >> 16);  // RNE
}

// ---- flat f32 -> bf16 (weights) -------------------------------------------
__global__ void f2bf_k(const float* __restrict__ in, unsigned short* __restrict__ out, int n) {
  int i = blockIdx.x * blockDim.x + threadIdx.x;
  if (i < n) out[i] = f2b(in[i]);
}

// ---- NCHW f32 -> NHWC bf16 ------------------------------------------------
__global__ void nchw2nhwc_k(const float* __restrict__ in, unsigned short* __restrict__ out,
                            int B, int C, int H, int W) {
  int i = blockIdx.x * blockDim.x + threadIdx.x;
  int total = B * C * H * W;
  if (i >= total) return;
  int c = i % C; int t = i / C;
  int x = t % W; t /= W;
  int y = t % H; int b = t / H;
  out[i] = f2b(in[(((size_t)b * C + c) * H + y) * W + x]);
}

// ---- WMMA GEMM:  Out[N,M] = X[N,K](bf16) * W[M,K]^T(bf16) + bias ----------
// One wave -> 16 rows x 64 cols (M must be a multiple of 64; K of 32).
// MODE: 0 = bf16 out, 1 = bf16 out + ReLU, 2 = f32 overwrite, 3 = f32 accum
template<int MODE>
__global__ __launch_bounds__(128) void gemm_k(const unsigned short* __restrict__ X,
                                              const unsigned short* __restrict__ Wb,
                                              const float* __restrict__ bias,
                                              void* __restrict__ Out,
                                              int N, int K, int M) {
  const int lane  = threadIdx.x & 31;
  const int wave  = blockIdx.x * (blockDim.x >> 5) + (threadIdx.x >> 5);
  const int grpM  = M >> 6;                        // 64-wide column groups
  const int total = (N >> 4) * grpM;
  if (wave >= total) return;                       // wave-uniform: EXEC stays all-ones
  const int ti = wave / grpM, tg = wave % grpM;
  const int l15 = lane & 15;
  const int hi  = lane >> 4;                       // half-wave select
  // A 16x32 bf16 layout: 16 contiguous K halves, hi half at K+16 (matches B)
  const size_t aBase = (size_t)(ti * 16 + l15) * K + (hi ? 16 : 0);
  // B 32x16 bf16 layout: lane = column, 16 contiguous K halves, hi half at K+16
  size_t bBase[4];
#pragma unroll
  for (int j = 0; j < 4; ++j)
    bBase[j] = (size_t)(tg * 64 + j * 16 + l15) * K + (hi ? 16 : 0);

  v8f acc[4];
#pragma unroll
  for (int j = 0; j < 4; ++j) acc[j] = (v8f){0.f, 0.f, 0.f, 0.f, 0.f, 0.f, 0.f, 0.f};

  for (int kb = 0; kb < K; kb += 32) {
    us8 a0 = *(const us8*)(X + aBase + kb);
    us8 a1 = *(const us8*)(X + aBase + kb + 8);
    U16cast A;
#pragma unroll
    for (int i = 0; i < 8; ++i) { A.u[i] = a0[i]; A.u[8 + i] = a1[i]; }
    U16cast Bv[4];
#pragma unroll
    for (int j = 0; j < 4; ++j) Bv[j].u = *(const us16*)(Wb + bBase[j] + kb);
#pragma unroll
    for (int j = 0; j < 4; ++j)
      acc[j] = __builtin_amdgcn_wmma_f32_16x16x32_bf16(false, A.b, false, Bv[j].b,
                                                       (short)0, acc[j], false, false);
  }

  const int rbase = ti * 16 + hi * 8;              // C/D: VGPR r -> row r (+8 for hi half)
#pragma unroll
  for (int j = 0; j < 4; ++j) {
    const int feat = tg * 64 + j * 16 + l15;
    const float bv = bias ? bias[feat] : 0.f;
#pragma unroll
    for (int r = 0; r < 8; ++r) {
      float v = acc[j][r] + bv;
      if (MODE == 1) v = v > 0.f ? v : 0.f;
      size_t idx = (size_t)(rbase + r) * M + feat;
      if (MODE <= 1)      ((unsigned short*)Out)[idx] = f2b(v);
      else if (MODE == 2) ((float*)Out)[idx] = v;
      else                ((float*)Out)[idx] += v;
    }
  }
}

// ---- fused 3x3-window local attention (8 heads x d=8) ---------------------
// mode 0: self (kv on same grid); 1: top-down (kv at y>>1,x>>1); 2: bottom-up
// (kv around 2y,2x on finer grid). pos bias added to keys even for OOB taps.
__global__ void attn_k(const unsigned short* __restrict__ q, int qstride, int qoff,
                       const unsigned short* __restrict__ kv, int kvstride, int koff, int voff,
                       const float* __restrict__ pos, unsigned short* __restrict__ o,
                       int B, int Hq, int Wq, int Hk, int Wk, int mode, float scale2) {
  int t = blockIdx.x * blockDim.x + threadIdx.x;
  int total = B * Hq * Wq * 8;
  if (t >= total) return;
  int head = t & 7; int pix = t >> 3;
  int x = pix % Wq; int r = pix / Wq;
  int y = r % Hq;   int b = r / Hq;
  int cy = (mode == 1) ? (y >> 1) : (mode == 2) ? (y << 1) : y;
  int cx = (mode == 1) ? (x >> 1) : (mode == 2) ? (x << 1) : x;

  float qv[8];
  const unsigned short* qp = q + (size_t)pix * qstride + qoff + head * 8;
#pragma unroll
  for (int i = 0; i < 8; ++i) qv[i] = b2f(qp[i]);

  float logits[9]; float mx = -1e30f;
#pragma unroll
  for (int n = 0; n < 9; ++n) {
    int ny = cy + n / 3 - 1, nx = cx + n % 3 - 1;
    bool inb = (ny >= 0) && (ny < Hk) && (nx >= 0) && (nx < Wk);
    const unsigned short* kp =
        kv + ((size_t)(b * Hk + (inb ? ny : 0)) * Wk + (inb ? nx : 0)) * kvstride + koff + head * 8;
    float dot = 0.f;
#pragma unroll
    for (int i = 0; i < 8; ++i) {
      float kvl = (inb ? b2f(kp[i]) : 0.f) + pos[(head * 8 + i) * 9 + n];
      dot += qv[i] * kvl;
    }
    logits[n] = dot * scale2;
    mx = fmaxf(mx, logits[n]);
  }
  float s = 0.f;
#pragma unroll
  for (int n = 0; n < 9; ++n) { logits[n] = __expf(logits[n] - mx); s += logits[n]; }
  float inv = 1.f / s;

  float out[8] = {0.f, 0.f, 0.f, 0.f, 0.f, 0.f, 0.f, 0.f};
#pragma unroll
  for (int n = 0; n < 9; ++n) {
    int ny = cy + n / 3 - 1, nx = cx + n % 3 - 1;
    if (ny >= 0 && ny < Hk && nx >= 0 && nx < Wk) {
      const unsigned short* vp =
          kv + ((size_t)(b * Hk + ny) * Wk + nx) * kvstride + voff + head * 8;
      float w = logits[n] * inv;
#pragma unroll
      for (int i = 0; i < 8; ++i) out[i] += w * b2f(vp[i]);
    }
  }
  unsigned short* op = o + (size_t)pix * 64 + head * 8;
#pragma unroll
  for (int i = 0; i < 8; ++i) op[i] = f2b(out[i]);
}

// ---- LN( x(bf16) + delta(f32) ) -> bf16 NHWC ------------------------------
__global__ void add_ln_k(const unsigned short* __restrict__ x, const float* __restrict__ d,
                         const float* __restrict__ g, const float* __restrict__ bta,
                         unsigned short* __restrict__ out, int Npix) {
  int p = blockIdx.x * blockDim.x + threadIdx.x;
  if (p >= Npix) return;
  const unsigned short* xp = x + (size_t)p * 64;
  const float* dp = d + (size_t)p * 64;
  float s = 0.f, s2 = 0.f;
  for (int f = 0; f < 64; ++f) { float v = b2f(xp[f]) + dp[f]; s += v; s2 += v * v; }
  float mu = s * (1.f / 64.f);
  float var = s2 * (1.f / 64.f) - mu * mu;
  float rs = rsqrtf(var + 1e-5f);
  for (int f = 0; f < 64; ++f) {
    float v = b2f(xp[f]) + dp[f];
    out[(size_t)p * 64 + f] = f2b((v - mu) * rs * g[f] + bta[f]);
  }
}

// ---- LN( x(bf16) + delta(f32) ) -> NCHW f32 (final output) ----------------
__global__ void add_ln_nchw_k(const unsigned short* __restrict__ x, const float* __restrict__ d,
                              const float* __restrict__ g, const float* __restrict__ bta,
                              float* __restrict__ out, int B, int H, int W) {
  int p = blockIdx.x * blockDim.x + threadIdx.x;
  int Npix = B * H * W;
  if (p >= Npix) return;
  int xw = p % W; int t = p / W;
  int y = t % H;  int b = t / H;
  const unsigned short* xp = x + (size_t)p * 64;
  const float* dp = d + (size_t)p * 64;
  float s = 0.f, s2 = 0.f;
  for (int f = 0; f < 64; ++f) { float v = b2f(xp[f]) + dp[f]; s += v; s2 += v * v; }
  float mu = s * (1.f / 64.f);
  float var = s2 * (1.f / 64.f) - mu * mu;
  float rs = rsqrtf(var + 1e-5f);
  for (int f = 0; f < 64; ++f) {
    float v = b2f(xp[f]) + dp[f];
    out[(((size_t)b * 64 + f) * H + y) * W + xw] = (v - mu) * rs * g[f] + bta[f];
  }
}

// ---------------------------------------------------------------------------
static inline int cdiv(int a, int b) { return (a + b - 1) / b; }

static void run_gemm(int mode, const unsigned short* X, const unsigned short* Wb,
                     const float* bias, void* Out, int N, int K, int M, hipStream_t s) {
  int tiles = (N / 16) * (M / 64);                 // 16x64 strip per wave
  dim3 grid(cdiv(tiles, 4)), block(128);
  switch (mode) {
    case 0: gemm_k<0><<<grid, block, 0, s>>>(X, Wb, bias, Out, N, K, M); break;
    case 1: gemm_k<1><<<grid, block, 0, s>>>(X, Wb, bias, Out, N, K, M); break;
    case 2: gemm_k<2><<<grid, block, 0, s>>>(X, Wb, bias, Out, N, K, M); break;
    default: gemm_k<3><<<grid, block, 0, s>>>(X, Wb, bias, Out, N, K, M); break;
  }
}

extern "C" void kernel_launch(void* const* d_in, const int* in_sizes, int n_in,
                              void* d_out, int out_size, void* d_ws, size_t ws_size,
                              hipStream_t stream) {
  (void)in_sizes; (void)n_in; (void)out_size; (void)ws_size;

  const int Bn = 4;
  const int Hs[3] = {128, 64, 32};
  int Np[3];
  for (int m = 0; m < 3; ++m) Np[m] = Bn * Hs[m] * Hs[m];
  const float scale2 = 0.35355339059327373f;  // (64/8)^(-0.25) squared

  // --- d_in leaf indices (jax pytree: sorted dict keys) --------------------
  // 0..2: map0..map2
  auto INP_B = [](int i) { return 3 + 2 * i; };
  auto INP_W = [](int i) { return 4 + 2 * i; };
  auto LN_B  = [](int i) { return 9 + 2 * i; };
  auto LN_G  = [](int i) { return 10 + 2 * i; };
  const int BU0 = 15;    // +7j: kv_b,kv_w,out_b,out_w,pos,q_b,q_w
  const int FFN0 = 29;   // +6i: in_b,in_w,ln_b,ln_g,out_b,out_w
  const int SELF0 = 47;  // +5i: out_b,out_w,pos,qkv_b,qkv_w
  const int TD0 = 62;    // +7j: kv_b,kv_w,out_b,out_w,pos,q_b,q_w
  auto F32 = [&](int i) { return (const float*)d_in[i]; };

  // --- workspace bump allocator -------------------------------------------
  char* base = (char*)d_ws;
  size_t off = 0;
  auto alloc = [&](size_t bytes) -> void* {
    void* p = base + off;
    off = (off + bytes + 255) & ~(size_t)255;
    return p;
  };
  auto cw = [&](int idx, int n) -> unsigned short* {  // f32 weight -> bf16 copy
    unsigned short* dst = (unsigned short*)alloc((size_t)n * 2);
    f2bf_k<<<cdiv(n, 256), 256, 0, stream>>>(F32(idx), dst, n);
    return dst;
  };

  unsigned short *w_inproj[3], *w_qkv[3], *w_sout[3], *w_fin[3], *w_fout[3];
  unsigned short *w_td_q[2], *w_td_kv[2], *w_td_o[2], *w_bu_q[2], *w_bu_kv[2], *w_bu_o[2];
  for (int m = 0; m < 3; ++m) {
    w_inproj[m] = cw(INP_W(m), 64 * 64);
    w_qkv[m]    = cw(SELF0 + 5 * m + 4, 192 * 64);
    w_sout[m]   = cw(SELF0 + 5 * m + 1, 64 * 64);
    w_fin[m]    = cw(FFN0 + 6 * m + 1, 256 * 64);
    w_fout[m]   = cw(FFN0 + 6 * m + 5, 64 * 256);
  }
  for (int j = 0; j < 2; ++j) {
    w_td_q[j]  = cw(TD0 + 7 * j + 6, 64 * 64);
    w_td_kv[j] = cw(TD0 + 7 * j + 1, 128 * 64);
    w_td_o[j]  = cw(TD0 + 7 * j + 3, 64 * 64);
    w_bu_q[j]  = cw(BU0 + 7 * j + 6, 64 * 64);
    w_bu_kv[j] = cw(BU0 + 7 * j + 1, 128 * 64);
    w_bu_o[j]  = cw(BU0 + 7 * j + 3, 64 * 64);
  }

  unsigned short* wx[3];
  float* wdelta[3];
  for (int m = 0; m < 3; ++m) wx[m] = (unsigned short*)alloc((size_t)Np[m] * 64 * 2);
  for (int m = 0; m < 3; ++m) wdelta[m] = (float*)alloc((size_t)Np[m] * 64 * 4);
  unsigned short* wT1 = (unsigned short*)alloc((size_t)Np[0] * 192 * 2);  // qkv / q+kv
  unsigned short* wO  = (unsigned short*)alloc((size_t)Np[0] * 64 * 2);   // attn out
  unsigned short* wH  = (unsigned short*)alloc((size_t)Np[0] * 256 * 2);  // ffn hidden
  unsigned short* wX1 = (unsigned short*)alloc((size_t)Np[0] * 64 * 2);   // staging / post-LN

  // --- input projection (NCHW f32 -> NHWC bf16 -> linear) ------------------
  for (int m = 0; m < 3; ++m) {
    int tot = Np[m] * 64;
    nchw2nhwc_k<<<cdiv(tot, 256), 256, 0, stream>>>(F32(m), wX1, Bn, 64, Hs[m], Hs[m]);
    run_gemm(0, wX1, w_inproj[m], F32(INP_B(m)), wx[m], Np[m], 64, 64, stream);
  }

  // --- self attention (delta overwrite) ------------------------------------
  for (int m = 0; m < 3; ++m) {
    run_gemm(0, wx[m], w_qkv[m], F32(SELF0 + 5 * m + 3), wT1, Np[m], 64, 192, stream);
    int th = Np[m] * 8;
    attn_k<<<cdiv(th, 256), 256, 0, stream>>>(wT1, 192, 0, wT1, 192, 64, 128,
                                              F32(SELF0 + 5 * m + 2), wO,
                                              Bn, Hs[m], Hs[m], Hs[m], Hs[m], 0, scale2);
    run_gemm(2, wO, w_sout[m], F32(SELF0 + 5 * m + 0), wdelta[m], Np[m], 64, 64, stream);
  }

  // --- top-down cross attention (fine m queries coarse m+1) ----------------
  for (int j = 0; j < 2; ++j) {
    int m = j;
    run_gemm(0, wx[m], w_td_q[j], F32(TD0 + 7 * j + 5), wT1, Np[m], 64, 64, stream);
    unsigned short* kvbuf = wT1 + (size_t)Np[m] * 64;
    run_gemm(0, wx[m + 1], w_td_kv[j], F32(TD0 + 7 * j + 0), kvbuf, Np[m + 1], 64, 128, stream);
    int th = Np[m] * 8;
    attn_k<<<cdiv(th, 256), 256, 0, stream>>>(wT1, 64, 0, kvbuf, 128, 0, 64,
                                              F32(TD0 + 7 * j + 4), wO,
                                              Bn, Hs[m], Hs[m], Hs[m + 1], Hs[m + 1], 1, scale2);
    run_gemm(3, wO, w_td_o[j], F32(TD0 + 7 * j + 2), wdelta[m], Np[m], 64, 64, stream);
  }

  // --- bottom-up cross attention (coarse m queries fine m-1) ---------------
  for (int j = 0; j < 2; ++j) {
    int m = j + 1;
    run_gemm(0, wx[m], w_bu_q[j], F32(BU0 + 7 * j + 5), wT1, Np[m], 64, 64, stream);
    unsigned short* kvbuf = wT1 + (size_t)Np[m] * 64;
    run_gemm(0, wx[m - 1], w_bu_kv[j], F32(BU0 + 7 * j + 0), kvbuf, Np[m - 1], 64, 128, stream);
    int th = Np[m] * 8;
    attn_k<<<cdiv(th, 256), 256, 0, stream>>>(wT1, 64, 0, kvbuf, 128, 0, 64,
                                              F32(BU0 + 7 * j + 4), wO,
                                              Bn, Hs[m], Hs[m], Hs[m - 1], Hs[m - 1], 2, scale2);
    run_gemm(3, wO, w_bu_o[j], F32(BU0 + 7 * j + 2), wdelta[m], Np[m], 64, 64, stream);
  }

  // --- residual + LN, FFN, residual + LN, emit NCHW f32 --------------------
  size_t ooff = 0;
  for (int m = 0; m < 3; ++m) {
    add_ln_k<<<cdiv(Np[m], 128), 128, 0, stream>>>(wx[m], wdelta[m], F32(LN_G(m)), F32(LN_B(m)),
                                                   wX1, Np[m]);
    run_gemm(1, wX1, w_fin[m], F32(FFN0 + 6 * m + 0), wH, Np[m], 64, 256, stream);      // ReLU
    run_gemm(2, wH, w_fout[m], F32(FFN0 + 6 * m + 4), wdelta[m], Np[m], 256, 64, stream);
    add_ln_nchw_k<<<cdiv(Np[m], 128), 128, 0, stream>>>(wX1, wdelta[m],
                                                        F32(FFN0 + 6 * m + 3), F32(FFN0 + 6 * m + 2),
                                                        (float*)d_out + ooff, Bn, Hs[m], Hs[m]);
    ooff += (size_t)Np[m] * 64;
  }
}
